// InterMultiplication_38792144617623
// MI455X (gfx1250) — matrix operand order repaired and checked by reading, not compiled, not run on hardware
//
#include <hip/hip_runtime.h>

#define NRES   768
#define CDIM   128
#define PADX   136     // padded LDS row (ushort): 272B = 68 dwords -> 4-bank rotation/row
#define CHUNKS 16
#define CHLEN  48      // 768 = 16 * 48
#define LINE_ELEMS (NRES * CDIM)  // 98304

typedef __attribute__((ext_vector_type(16))) __bf16 bf16x16;
typedef __attribute__((ext_vector_type(8)))  float  f32x8;

#define WMMA_BF16(A, B, C) \
  __builtin_amdgcn_wmma_f32_16x16x32_bf16(false, (A), false, (B), (short)0, (C), false, false)

__device__ __forceinline__ unsigned short f32_to_bf16(float f) {
  unsigned int u = __float_as_uint(f);
  u += 0x7FFFu + ((u >> 16) & 1u);   // round-to-nearest-even
  return (unsigned short)(u >> 16);
}

// reduce across a 16-lane half of a wave32 (xor 1,2,4,8 stays within the half)
__device__ __forceinline__ float half16_reduce(float v) {
  v += __shfl_xor(v, 1, 32);
  v += __shfl_xor(v, 2, 32);
  v += __shfl_xor(v, 4, 32);
  v += __shfl_xor(v, 8, 32);
  return v;
}

// A-matrix 16x32 bf16 fragment from an LDS row (lane = M = lane&15).
// entries e -> K = k0 + 16*khalf + e  (16 consecutive ushorts)
__device__ __forceinline__ bf16x16 load_a_frag_lds(const unsigned short* xrow, int k0, int khalf) {
  union { unsigned int u[8]; bf16x16 v; } f;
  const unsigned int* p0 = (const unsigned int*)(xrow + k0 + 16 * khalf);
  const unsigned int* p1 = (const unsigned int*)(xrow + k0 + 16 * khalf + 8);
#pragma unroll
  for (int i = 0; i < 4; ++i) { f.u[i] = p0[i]; f.u[4 + i] = p1[i]; }
  return f.v;
}

// B-matrix 32x16 bf16 fragment from transposed-weights LDS (Wt[n][k]); lane = N = lane&15.
// entries e -> K = k0 + 16*khalf + e  (16 consecutive ushorts)
__device__ __forceinline__ bf16x16 load_b_frag_lds(const unsigned short* wrow, int k0, int khalf) {
  union { unsigned int u[8]; bf16x16 v; } f;
  const unsigned int* p = (const unsigned int*)(wrow + k0 + 16 * khalf);
#pragma unroll
  for (int i = 0; i < 8; ++i) f.u[i] = p[i];
  return f.v;
}

// B fragment straight from a global f32 [K][N] weight (hoisted once per block).
__device__ __forceinline__ bf16x16 load_b_frag_global(const float* __restrict__ W, int n, int kbase) {
  union { unsigned short s[16]; bf16x16 v; } f;
#pragma unroll
  for (int e = 0; e < 16; ++e) f.s[e] = f32_to_bf16(W[(size_t)(kbase + e) * CDIM + n]);
  return f.v;
}

// ---------------------------------------------------------------------------
// K1: fused input-LN + (proj, gate) GEMMs + sigmoid gate (+mask) + partial sum
//     over the a-axis chunk. grid = (48 b-tiles, 16 chunks), block = 256 (8 waves).
//     Double-buffered bf16 tile: one barrier/iter, LN(i+1) co-executes with WMMA(i).
// ---------------------------------------------------------------------------
__global__ __launch_bounds__(256) void k_ln_proj_gate_reduce(
    const float* __restrict__ act, const float* __restrict__ mask,
    const float* __restrict__ ln_g, const float* __restrict__ ln_b,
    const float* __restrict__ Wp, const float* __restrict__ bp,
    const float* __restrict__ Wg, const float* __restrict__ bg,
    float* __restrict__ partial, int use_mask) {
  __shared__ unsigned short Xn[2][16][PADX];
  __shared__ float sg[CDIM], sb[CDIM];

  const int tid   = threadIdx.x;
  const int lane  = tid & 31;
  const int wv    = tid >> 5;
  const int khalf = lane >> 4;
  const int ncol  = lane & 15;
  const int b0    = blockIdx.x * 16;
  const int a_begin = blockIdx.y * CHLEN;

  if (tid < CDIM) { sg[tid] = ln_g[tid]; sb[tid] = ln_b[tid]; }

  const int n0 = wv * 16;                  // this wave's N-tile
  const float bias_p = bp[n0 + ncol];
  const float bias_g = bg[n0 + ncol];

  // Hoist all B fragments (both weight matrices) in registers for the a-loop.
  bf16x16 B1[4], B2[4];
#pragma unroll
  for (int kk = 0; kk < 4; ++kk) {
    const int kbase = kk * 32 + 16 * khalf;
    B1[kk] = load_b_frag_global(Wp, n0 + ncol, kbase);
    B2[kk] = load_b_frag_global(Wg, n0 + ncol, kbase);
  }

  const int ln_row = 2 * wv + khalf;       // this half-wave's LN row
  const int cb     = ncol * 8;             // this lane's 8 channels

  // LayerNorm one 16x128 tile (2 rows/wave, half-wave/row) into buffer `buf`.
  auto ln_tile = [&](int a, int buf) {
    const float* src = act + ((size_t)a * NRES + (b0 + ln_row)) * CDIM + cb;
    const float4 v0 = ((const float4*)src)[0];
    const float4 v1 = ((const float4*)src)[1];
    float x[8] = {v0.x, v0.y, v0.z, v0.w, v1.x, v1.y, v1.z, v1.w};
    float s = 0.f, ss = 0.f;
#pragma unroll
    for (int i = 0; i < 8; ++i) { s += x[i]; ss += x[i] * x[i]; }
    s = half16_reduce(s); ss = half16_reduce(ss);
    const float mean = s * (1.f / CDIM);
    const float var  = ss * (1.f / CDIM) - mean * mean;
    const float rstd = __builtin_amdgcn_rsqf(var + 1e-5f);   // single v_rsq_f32
#pragma unroll
    for (int i = 0; i < 8; ++i) {
      const float y = (x[i] - mean) * rstd * sg[cb + i] + sb[cb + i];
      Xn[buf][ln_row][cb + i] = f32_to_bf16(y);
    }
  };

  f32x8 acc = {0.f, 0.f, 0.f, 0.f, 0.f, 0.f, 0.f, 0.f};
  __syncthreads();                         // sg/sb visible
  ln_tile(a_begin, 0);                     // prologue fill of buffer 0

  for (int ai = 0; ai < CHLEN; ++ai) {
    __syncthreads();                       // Xn[cur] fully written by all waves
    const int cur = ai & 1;
    const int a   = a_begin + ai;

    // Keep L2 ahead of the stream (gfx1250 global_prefetch_b8).
    if (ai + 2 < CHLEN)
      __builtin_prefetch(act + ((size_t)(a + 2) * NRES + (b0 + ln_row)) * CDIM + cb, 0, 1);

    // Software pipeline: LN of next tile into the other buffer while this
    // iteration's WMMAs (TRANS-tracked) co-execute.
    if (ai + 1 < CHLEN) ln_tile(a + 1, cur ^ 1);

    // Mask values for this lane's 8 output rows: contiguous broadcast loads.
    float mv[8] = {1.f, 1.f, 1.f, 1.f, 1.f, 1.f, 1.f, 1.f};
    if (use_mask) {
      const float* mp = mask + (size_t)a * NRES + b0 + 8 * khalf;
      const float4 m0 = ((const float4*)mp)[0];
      const float4 m1 = ((const float4*)mp)[1];
      mv[0] = m0.x; mv[1] = m0.y; mv[2] = m0.z; mv[3] = m0.w;
      mv[4] = m1.x; mv[5] = m1.y; mv[6] = m1.z; mv[7] = m1.w;
    }

    f32x8 p = {0.f, 0.f, 0.f, 0.f, 0.f, 0.f, 0.f, 0.f};
    f32x8 g = {0.f, 0.f, 0.f, 0.f, 0.f, 0.f, 0.f, 0.f};
#pragma unroll
    for (int kk = 0; kk < 4; ++kk) {
      const bf16x16 A = load_a_frag_lds(&Xn[cur][ncol][0], kk * 32, khalf);
      p = WMMA_BF16(A, B1[kk], p);
      g = WMMA_BF16(A, B2[kk], g);
    }
#pragma unroll
    for (int e = 0; e < 8; ++e) {
      // sigmoid via single v_rcp_f32 (~1 ulp) instead of IEEE divide sequence
      const float sgm = __builtin_amdgcn_rcpf(1.f + __expf(-(g[e] + bias_g)));
      acc[e] += (p[e] + bias_p) * sgm * mv[e];
    }
  }

#pragma unroll
  for (int e = 0; e < 8; ++e) {
    const int row = khalf ? (8 + e) : e;
    partial[((size_t)blockIdx.y * NRES + (b0 + row)) * CDIM + n0 + ncol] = acc[e];
  }
}

// ---------------------------------------------------------------------------
// K2: deterministic fixed-order chunk reduction -> line tensors [768,128]
// ---------------------------------------------------------------------------
__global__ __launch_bounds__(256) void k_reduce(const float* __restrict__ partial,
                                                float* __restrict__ line) {
  const int i = blockIdx.x * 256 + threadIdx.x;
  float s = 0.f;
#pragma unroll
  for (int c = 0; c < CHUNKS; ++c) s += partial[(size_t)c * LINE_ELEMS + i];
  line[i] = s;
}

// ---------------------------------------------------------------------------
// K3: outer product + center LN + output projection + bias.
//     grid = 768*768/128 blocks; each block owns 128 consecutive (i,j) rows
//     (768 % 128 == 0 so i is constant per block). block = 256 (8 waves),
//     wave w owns rows 16w..16w+15 and computes their full 128-wide output.
// ---------------------------------------------------------------------------
__global__ __launch_bounds__(256) void k_outer_ln_proj(
    const float* __restrict__ lineL, const float* __restrict__ lineR,
    const float* __restrict__ gC, const float* __restrict__ bC,
    const float* __restrict__ Wo, const float* __restrict__ bo,
    float* __restrict__ out) {
  extern __shared__ unsigned char smem[];
  unsigned short* WtO = (unsigned short*)smem;        // [128][PADX] transposed w_out (bf16)
  unsigned short* Xn  = WtO + CDIM * PADX;            // [128][PADX] normalized rows (bf16)
  float* ll   = (float*)(Xn + CDIM * PADX);           // left_line[i][:]
  float* g_s  = ll + CDIM;
  float* b_s  = g_s + CDIM;
  float* bo_s = b_s + CDIM;

  const int tid   = threadIdx.x;
  const int lane  = tid & 31;
  const int wv    = tid >> 5;
  const int khalf = lane >> 4;
  const int ncol  = lane & 15;
  const int p0    = blockIdx.x * 128;
  const int i0    = p0 / NRES;
  const int j0    = p0 - i0 * NRES;

  for (int idx = tid; idx < CDIM * CDIM; idx += 256) {
    const int k = idx >> 7, n = idx & (CDIM - 1);
    WtO[n * PADX + k] = f32_to_bf16(Wo[idx]);         // transpose: Wt[n][k] = W[k][n]
  }
  if (tid < CDIM) {
    ll[tid]   = lineL[(size_t)i0 * CDIM + tid];
    g_s[tid]  = gC[tid];
    b_s[tid]  = bC[tid];
    bo_s[tid] = bo[tid];
  }
  __syncthreads();

  // Per-wave: LN of 16 outer-product rows (2 rows per pass, half-wave per row).
#pragma unroll
  for (int mm = 0; mm < 8; ++mm) {
    const int m  = 16 * wv + 2 * mm + khalf;          // local row 0..127
    const int j  = j0 + m;
    const int cb = ncol * 8;
    const float* rr = lineR + (size_t)j * CDIM + cb;
    const float4 v0 = ((const float4*)rr)[0];
    const float4 v1 = ((const float4*)rr)[1];
    float x[8] = {v0.x, v0.y, v0.z, v0.w, v1.x, v1.y, v1.z, v1.w};
#pragma unroll
    for (int i = 0; i < 8; ++i) x[i] *= ll[cb + i];   // outer product
    float s = 0.f, ss = 0.f;
#pragma unroll
    for (int i = 0; i < 8; ++i) { s += x[i]; ss += x[i] * x[i]; }
    s = half16_reduce(s); ss = half16_reduce(ss);
    const float mean = s * (1.f / CDIM);
    const float var  = ss * (1.f / CDIM) - mean * mean;
    const float rstd = __builtin_amdgcn_rsqf(var + 1e-5f);
#pragma unroll
    for (int i = 0; i < 8; ++i) {
      const float y = (x[i] - mean) * rstd * g_s[cb + i] + b_s[cb + i];
      Xn[m * PADX + cb + i] = f32_to_bf16(y);
    }
  }
  // Xn rows for this wave are produced and consumed by this wave only: no barrier.

  bf16x16 A[4];
#pragma unroll
  for (int kk = 0; kk < 4; ++kk)
    A[kk] = load_a_frag_lds(Xn + (16 * wv + ncol) * PADX, kk * 32, khalf);

#pragma unroll
  for (int t = 0; t < 8; ++t) {                       // 8 N-tiles of 16
    f32x8 acc = {0.f, 0.f, 0.f, 0.f, 0.f, 0.f, 0.f, 0.f};
#pragma unroll
    for (int kk = 0; kk < 4; ++kk) {
      const bf16x16 B = load_b_frag_lds(WtO + (16 * t + ncol) * PADX, kk * 32, khalf);
      acc = WMMA_BF16(A[kk], B, acc);
    }
    const float bias = bo_s[16 * t + ncol];
#pragma unroll
    for (int e = 0; e < 8; ++e) {
      const int row = khalf ? (8 + e) : e;
      out[((size_t)(p0 + 16 * wv + row)) * CDIM + 16 * t + ncol] = acc[e] + bias;
    }
  }
}

// ---------------------------------------------------------------------------
extern "C" void kernel_launch(void* const* d_in, const int* in_sizes, int n_in,
                              void* d_out, int out_size, void* d_ws, size_t ws_size,
                              hipStream_t stream) {
  (void)in_sizes; (void)n_in; (void)out_size; (void)ws_size;
  const float* act_s   = (const float*)d_in[0];
  const float* act_r   = (const float*)d_in[1];
  const float* mask_r  = (const float*)d_in[2];
  const float* ln_in_g = (const float*)d_in[3];
  const float* ln_in_b = (const float*)d_in[4];
  const float* w_lp    = (const float*)d_in[5];
  const float* b_lp    = (const float*)d_in[6];
  const float* w_rp    = (const float*)d_in[7];
  const float* b_rp    = (const float*)d_in[8];
  const float* w_lg    = (const float*)d_in[9];
  const float* b_lg    = (const float*)d_in[10];
  const float* w_rg    = (const float*)d_in[11];
  const float* b_rg    = (const float*)d_in[12];
  const float* ln_c_g  = (const float*)d_in[13];
  const float* ln_c_b  = (const float*)d_in[14];
  const float* w_out   = (const float*)d_in[15];
  const float* b_out   = (const float*)d_in[16];
  float* out = (float*)d_out;

  float* ws    = (float*)d_ws;
  float* partL = ws;                                     // [CHUNKS][768][128]
  float* partR = partL + (size_t)CHUNKS * LINE_ELEMS;    // [CHUNKS][768][128]
  float* lineL = partR + (size_t)CHUNKS * LINE_ELEMS;    // [768][128]
  float* lineR = lineL + LINE_ELEMS;                     // [768][128]

  const dim3 gA(NRES / 16, CHUNKS);
  k_ln_proj_gate_reduce<<<gA, 256, 0, stream>>>(act_s, nullptr, ln_in_g, ln_in_b,
                                                w_lp, b_lp, w_lg, b_lg, partL, 0);
  k_ln_proj_gate_reduce<<<gA, 256, 0, stream>>>(act_r, mask_r, ln_in_g, ln_in_b,
                                                w_rp, b_rp, w_rg, b_rg, partR, 1);
  k_reduce<<<LINE_ELEMS / 256, 256, 0, stream>>>(partL, lineL);
  k_reduce<<<LINE_ELEMS / 256, 256, 0, stream>>>(partR, lineR);

  const size_t smemC = (size_t)2 * CDIM * PADX * sizeof(unsigned short) +
                       (size_t)4 * CDIM * sizeof(float);
  k_outer_ln_proj<<<(NRES * NRES) / 128, 256, smemC, stream>>>(
      lineL, lineR, ln_c_g, ln_c_b, w_out, b_out, out);
}
